// Attention_1580547965036
// MI455X (gfx1250) — matrix operand order repaired and checked by reading, not compiled, not run on hardware
//
#include <hip/hip_runtime.h>
#include <hip/hip_bf16.h>
#include <math.h>

// Problem constants (from reference)
#define SEQ   2048
#define BATCH 32
#define HDIM  1024
#define NLAY  2
#define LROW  1032   // bf16 elems per LDS row: 1024 + 8 pad -> row stride 2064B = 516 dwords (4-bank skew)

typedef __attribute__((ext_vector_type(16))) __bf16 v16bf;
typedef __attribute__((ext_vector_type(8)))  float  v8f;

union Frag32 { uint4 u[2]; v16bf v; };  // 32-byte bf16 WMMA fragment

static __device__ __forceinline__ unsigned short f32_to_bf16(float f) {
  unsigned int u = __float_as_uint(f);
  u += 0x7FFFu + ((u >> 16) & 1u);          // round-to-nearest-even
  return (unsigned short)(u >> 16);
}

// Native CDNA5 v_tanh_f32 (single TRANS op, co-executes with WMMA) instead of
// the branchy libm tanhf expansion.
static __device__ __forceinline__ float fast_tanh(float x) {
#if __has_builtin(__builtin_amdgcn_tanhf)
  return __builtin_amdgcn_tanhf(x);
#else
  float r;
  // v_nop covers the TRANS-op "1 instruction before result use" hazard, since
  // the compiler cannot schedule around the asm's internal dependency.
  asm("v_tanh_f32 %0, %1\n\tv_nop" : "=v"(r) : "v"(x));
  return r;
#endif
}

// ---------------------------------------------------------------------------
// Kernel 0a: convert W2 = attn_w[:, H:2H] (fp32) -> bf16, row-major [H][H]
// ---------------------------------------------------------------------------
__global__ __launch_bounds__(256) void convert_w2_kernel(
    const float* __restrict__ attn_w, unsigned short* __restrict__ w2bf) {
  int idx = blockIdx.x * 256 + threadIdx.x;      // 0 .. H*H-1
  int h = idx >> 10;
  int k = idx & (HDIM - 1);
  w2bf[idx] = f32_to_bf16(attn_w[(size_t)h * (2 * HDIM) + HDIM + k]);
}

// ---------------------------------------------------------------------------
// Kernel 0b: hb[b][h] = dot(W1[h,:], hidden_last[b,:]) + attn_b[h]   (fp32)
// ---------------------------------------------------------------------------
__global__ __launch_bounds__(256) void hb_kernel(
    const float* __restrict__ attn_w, const float* __restrict__ hid_last,
    const float* __restrict__ attn_b, float* __restrict__ hb) {
  int h = blockIdx.x * 256 + threadIdx.x;        // 0..1023
  int b = blockIdx.y;                            // 0..31
  const float4* wrow = (const float4*)(attn_w + (size_t)h * (2 * HDIM));
  const float4* xrow = (const float4*)(hid_last + (size_t)b * HDIM);
  float acc = attn_b[h];
#pragma unroll 4
  for (int k = 0; k < HDIM / 4; ++k) {
    float4 w = wrow[k];
    float4 x = xrow[k];
    acc += w.x * x.x + w.y * x.y + w.z * x.z + w.w * x.w;
  }
  hb[(size_t)b * HDIM + h] = acc;
}

// ---------------------------------------------------------------------------
// Kernel 1: fused  scores[b][s] = sum_h v[h] * tanh( enc[s,b,:]·W2[h,:] + hb[b][h] )
// One workgroup per s. 8 waves x 8 h-tiles. bf16 WMMA 16x16x32, K-loop over 1024.
// ---------------------------------------------------------------------------
__global__ __launch_bounds__(256) void attn_score_kernel(
    const float* __restrict__ enc,            // [S][B][H] fp32
    const unsigned short* __restrict__ w2,    // [H][H]   bf16
    const float* __restrict__ hb,             // [B][H]   fp32
    const float* __restrict__ vvec,           // [H]      fp32
    float* __restrict__ scores) {             // [B][S]   fp32
  __shared__ unsigned short Abuf[BATCH * LROW];   // 66 KB (<< 320 KB WGP LDS)
  __shared__ float scoreLDS[BATCH];

  const int s   = blockIdx.x;
  const int tid = threadIdx.x;
  if (tid < BATCH) scoreLDS[tid] = 0.0f;

  // ---- Stage A-block: enc[s] (32 rows x 1024 fp32) -> bf16 in LDS, padded rows
  const float4* src4 = (const float4*)(enc + (size_t)s * BATCH * HDIM);
  for (int i = tid; i < (BATCH * HDIM) / 4; i += 256) {
    float4 f = src4[i];
    int e   = i << 2;
    int row = e >> 10;
    int col = e & (HDIM - 1);
    uint2 p;
    p.x = (unsigned)f32_to_bf16(f.x) | ((unsigned)f32_to_bf16(f.y) << 16);
    p.y = (unsigned)f32_to_bf16(f.z) | ((unsigned)f32_to_bf16(f.w) << 16);
    *(uint2*)&Abuf[row * LROW + col] = p;         // ds_store_b64, 8B aligned
  }
  __syncthreads();

  const int lane   = tid & 31;
  const int wave   = tid >> 5;
  const int laneN  = lane & 15;   // column n of the tile / B-matrix column
  const int laneHi = lane >> 4;   // K-half selector per ISA fragment layout

  // A fragment base addresses (ISA layout: lanes 0-15 K{0-7,16-23}, lanes 16-31 K{8-15,24-31})
  const unsigned short* a0base = &Abuf[laneN * LROW + laneHi * 8];          // rows b=0..15
  const unsigned short* a1base = &Abuf[(16 + laneN) * LROW + laneHi * 8];   // rows b=16..31

  float part0[8] = {0.f, 0.f, 0.f, 0.f, 0.f, 0.f, 0.f, 0.f};
  float part1[8] = {0.f, 0.f, 0.f, 0.f, 0.f, 0.f, 0.f, 0.f};

  for (int t = 0; t < 8; ++t) {
    const int h0 = wave * 128 + t * 16;
    // B column n = W2 row (h0+n); lanes 0-15 K{0-7,16-23}, lanes 16-31 K{8-15,24-31}
    const unsigned short* bptr = w2 + ((size_t)(h0 + laneN) << 10) + (laneHi << 3);

    v8f c0 = {0.f, 0.f, 0.f, 0.f, 0.f, 0.f, 0.f, 0.f};
    v8f c1 = {0.f, 0.f, 0.f, 0.f, 0.f, 0.f, 0.f, 0.f};

#pragma unroll 4
    for (int kc = 0; kc < HDIM; kc += 32) {
      Frag32 bf, a0, a1;
      bf.u[0] = *(const uint4*)(bptr + kc);        // global_load_b128
      bf.u[1] = *(const uint4*)(bptr + kc + 16);
      a0.u[0] = *(const uint4*)(a0base + kc);      // ds_load_b128 (conflict-free via row skew)
      a0.u[1] = *(const uint4*)(a0base + kc + 16);
      a1.u[0] = *(const uint4*)(a1base + kc);
      a1.u[1] = *(const uint4*)(a1base + kc + 16);
      c0 = __builtin_amdgcn_wmma_f32_16x16x32_bf16(false, a0.v, false, bf.v,
                                                   (short)0, c0, false, false);
      c1 = __builtin_amdgcn_wmma_f32_16x16x32_bf16(false, a1.v, false, bf.v,
                                                   (short)0, c1, false, false);
    }

    // Epilogue: element (m, n): lane holds n = laneN, m = i + 8*laneHi; b = m (tile0) / m+16 (tile1)
    const int h = h0 + laneN;
    const float vh = vvec[h];
    const float* hb_h = hb + h;
#pragma unroll
    for (int i = 0; i < 8; ++i) {
      int m = (laneHi << 3) + i;
      part0[i] += vh * fast_tanh(c0[i] + hb_h[(size_t)m * HDIM]);
      part1[i] += vh * fast_tanh(c1[i] + hb_h[(size_t)(m + 16) * HDIM]);
    }
  }

  // Reduce over n within each 16-lane half (xor masks 8..1 stay inside the half)
#pragma unroll
  for (int i = 0; i < 8; ++i) {
    float p0 = part0[i], p1 = part1[i];
#pragma unroll
    for (int off = 8; off >= 1; off >>= 1) {
      p0 += __shfl_xor(p0, off, 32);
      p1 += __shfl_xor(p1, off, 32);
    }
    if (laneN == 0) {
      atomicAdd(&scoreLDS[(laneHi << 3) + i], p0);        // b = m
      atomicAdd(&scoreLDS[16 + (laneHi << 3) + i], p1);   // b = m + 16
    }
  }
  __syncthreads();
  if (tid < BATCH) scores[(size_t)tid * SEQ + s] = scoreLDS[tid];
}

// ---------------------------------------------------------------------------
// Kernel 2: row softmax over S per batch element
// ---------------------------------------------------------------------------
__global__ __launch_bounds__(256) void softmax_kernel(
    const float* __restrict__ scores, float* __restrict__ out) {
  __shared__ float red[8];
  const int b = blockIdx.x;
  const int tid = threadIdx.x;
  const int lane = tid & 31, wave = tid >> 5;
  const float* row = scores + (size_t)b * SEQ;

  float vals[8];
  float m = -1e30f;
#pragma unroll
  for (int i = 0; i < 8; ++i) {
    vals[i] = row[tid + i * 256];
    m = fmaxf(m, vals[i]);
  }
#pragma unroll
  for (int off = 16; off >= 1; off >>= 1) m = fmaxf(m, __shfl_xor(m, off, 32));
  if (lane == 0) red[wave] = m;
  __syncthreads();
  float mm = red[0];
#pragma unroll
  for (int j = 1; j < 8; ++j) mm = fmaxf(mm, red[j]);
  __syncthreads();

  float sum = 0.f;
#pragma unroll
  for (int i = 0; i < 8; ++i) {
    vals[i] = __expf(vals[i] - mm);
    sum += vals[i];
  }
#pragma unroll
  for (int off = 16; off >= 1; off >>= 1) sum += __shfl_xor(sum, off, 32);
  if (lane == 0) red[wave] = sum;
  __syncthreads();
  float total = 0.f;
#pragma unroll
  for (int j = 0; j < 8; ++j) total += red[j];
  float inv = 1.0f / total;
#pragma unroll
  for (int i = 0; i < 8; ++i) out[(size_t)b * SEQ + tid + i * 256] = vals[i] * inv;
}

// ---------------------------------------------------------------------------
// Launch
// ---------------------------------------------------------------------------
extern "C" void kernel_launch(void* const* d_in, const int* in_sizes, int n_in,
                              void* d_out, int out_size, void* d_ws, size_t ws_size,
                              hipStream_t stream) {
  const float* hidden = (const float*)d_in[0];   // [NL, B, H]
  const float* enc    = (const float*)d_in[1];   // [S, B, H]
  const float* attn_w = (const float*)d_in[2];   // [H, 2H]
  const float* attn_b = (const float*)d_in[3];   // [H]
  const float* vvec   = (const float*)d_in[4];   // [H]
  float* out = (float*)d_out;                    // [B, S]

  char* ws = (char*)d_ws;
  unsigned short* w2bf = (unsigned short*)ws;                                    // 2 MB
  float* hb     = (float*)(ws + (size_t)HDIM * HDIM * 2);                        // 128 KB
  float* scores = (float*)(ws + (size_t)HDIM * HDIM * 2 + (size_t)BATCH * HDIM * 4); // 256 KB

  const float* hid_last = hidden + (size_t)(NLAY - 1) * BATCH * HDIM;

  convert_w2_kernel<<<dim3((HDIM * HDIM) / 256), dim3(256), 0, stream>>>(attn_w, w2bf);
  hb_kernel<<<dim3(HDIM / 256, BATCH), dim3(256), 0, stream>>>(attn_w, hid_last, attn_b, hb);
  attn_score_kernel<<<dim3(SEQ), dim3(256), 0, stream>>>(enc, w2bf, hb, vvec, scores);
  softmax_kernel<<<dim3(BATCH), dim3(256), 0, stream>>>(scores, out);
}
